// EagleDraftHead_22419729285275
// MI455X (gfx1250) — compile-verified, matrix operand order checked by reading, not run on hardware
//
#include <hip/hip_runtime.h>
#include <hip/hip_bf16.h>

// ---------------------------------------------------------------------------
// Types
// ---------------------------------------------------------------------------
typedef __bf16 bf16_t;
typedef __bf16 v16bf __attribute__((ext_vector_type(16)));
typedef __bf16 v8bf  __attribute__((ext_vector_type(8)));
typedef float  v8f   __attribute__((ext_vector_type(8)));

__device__ __forceinline__ bf16_t f2b(float f) {
    unsigned u = __builtin_bit_cast(unsigned, f);
    unsigned r = (u + 0x7FFFu + ((u >> 16) & 1u)) >> 16;   // round-to-nearest-even
    unsigned short s = (unsigned short)r;
    return __builtin_bit_cast(bf16_t, s);
}
__device__ __forceinline__ float b2f(bf16_t b) {
    unsigned short s = __builtin_bit_cast(unsigned short, b);
    unsigned u = ((unsigned)s) << 16;
    return __builtin_bit_cast(float, u);
}

// ---------------------------------------------------------------------------
// Problem constants
// ---------------------------------------------------------------------------
constexpr int CB = 2, CS = 2048, CHS = 4096, CNH = 32, CHD = 128, CINTER = 11008;
constexpr int CM = CB * CS;               // 4096 tokens

// ---------------------------------------------------------------------------
// WMMA fragment load: 16-bit A/B layout (16x32 tile slice per lane):
//   lane row = lane&15 ; K chunks: (lane>>4)*8 .. +7  and  +16 .. +23
// ---------------------------------------------------------------------------
__device__ __forceinline__ v16bf load_frag(const bf16_t* p) {
    v8bf lo = *(const v8bf*)(p);
    v8bf hi = *(const v8bf*)(p + 16);
    v16bf r;
#pragma unroll
    for (int i = 0; i < 8; ++i) { r[i] = lo[i]; r[i + 8] = hi[i]; }
    return r;
}
__device__ __forceinline__ void store_c(float* p, float v)  { *p = v; }
__device__ __forceinline__ void store_c(bf16_t* p, float v) { *p = f2b(v); }

// ---------------------------------------------------------------------------
// fp32 -> bf16 bulk conversion (8 elems/thread, 16B vector stores)
// ---------------------------------------------------------------------------
__global__ __launch_bounds__(256) void eagle_f32_to_bf16(
    const float* __restrict__ src, bf16_t* __restrict__ dst)
{
    const size_t i = ((size_t)blockIdx.x * 256 + threadIdx.x) * 8;
    const float4 a = *(const float4*)(src + i);
    const float4 b = *(const float4*)(src + i + 4);
    v8bf o;
    o[0] = f2b(a.x); o[1] = f2b(a.y); o[2] = f2b(a.z); o[3] = f2b(a.w);
    o[4] = f2b(b.x); o[5] = f2b(b.y); o[6] = f2b(b.z); o[7] = f2b(b.w);
    *(v8bf*)(dst + i) = o;
}

// ---------------------------------------------------------------------------
// Tiled WMMA GEMM (pure bf16 inputs):  C[M,N] = A[M,K] * B[N,K]^T (+Cin)(+bias)
// Block: 256 thr (8 waves). Block tile 128x256, wave tile 64x64,
// 16 x v_wmma_f32_16x16x32_bf16 per K=32 step, zero VALU conversions in loop.
// VT: store output transposed per-head as [B,H,D,S] (used for the V proj).
// ---------------------------------------------------------------------------
template <typename TC, bool ADDC, bool BIAS, bool VT>
__global__ __launch_bounds__(256) void gemm_wmma(
    const bf16_t* __restrict__ A, const bf16_t* __restrict__ Bw,
    const float* Cin, const float* __restrict__ bias,
    TC* C, int M, int N, int K)
{
    const int wave = threadIdx.x >> 5;
    const int lane = threadIdx.x & 31;
    const int lrow = lane & 15;
    const int koff = (lane >> 4) << 3;
    const int wm = blockIdx.x * 128 + (wave >> 2) * 64;
    const int wn = blockIdx.y * 256 + (wave & 3) * 64;

    v8f acc[4][4];
#pragma unroll
    for (int i = 0; i < 4; ++i)
#pragma unroll
        for (int j = 0; j < 4; ++j)
#pragma unroll
            for (int e = 0; e < 8; ++e) acc[i][j][e] = 0.0f;

    const bf16_t* aptr = A  + (size_t)(wm + lrow) * K + koff;
    const bf16_t* bptr = Bw + (size_t)(wn + lrow) * K + koff;

    for (int k = 0; k < K; k += 32) {
        // prefetch the streamed tiles two K-steps ahead
        __builtin_prefetch(bptr + k + 64, 0, 1);
        __builtin_prefetch(aptr + k + 64, 0, 1);
        v16bf af[4], bfr[4];
#pragma unroll
        for (int t = 0; t < 4; ++t) af[t]  = load_frag(aptr + (size_t)t * 16 * K + k);
#pragma unroll
        for (int t = 0; t < 4; ++t) bfr[t] = load_frag(bptr + (size_t)t * 16 * K + k);
#pragma unroll
        for (int i = 0; i < 4; ++i)
#pragma unroll
            for (int j = 0; j < 4; ++j)
                acc[i][j] = __builtin_amdgcn_wmma_f32_16x16x32_bf16(
                    false, af[i], false, bfr[j], (short)0, acc[i][j], false, false);
    }

    // Epilogue. C layout: VGPR r -> row = r + 8*(lane>>4), col = lane&15.
#pragma unroll
    for (int i = 0; i < 4; ++i) {
#pragma unroll
        for (int r = 0; r < 8; ++r) {
            const int row = wm + i * 16 + r + ((lane >> 4) << 3);
#pragma unroll
            for (int j = 0; j < 4; ++j) {
                const int col = wn + j * 16 + lrow;
                float v = acc[i][j][r];
                if constexpr (ADDC) v += Cin[(size_t)row * N + col];
                if constexpr (BIAS) v += bias[col];
                if constexpr (VT) {
                    // row = b*S+s ; col = h*HD+d  ->  [(b*NH+h)*HD+d][s]
                    const int hh = col >> 7, d = col & (CHD - 1);
                    const int bb = row >> 11, s = row & (CS - 1);
                    store_c(&C[(((size_t)((bb * CNH + hh) * CHD + d)) * CS) + s], v);
                } else {
                    store_c(&C[(size_t)row * N + col], v);
                }
            }
        }
    }
}

// ---------------------------------------------------------------------------
// Embedding gather + concat -> bf16 Xcat[M, 2*HS]
// ---------------------------------------------------------------------------
__global__ __launch_bounds__(256) void eagle_embed_concat(
    const float* __restrict__ hid, const int* __restrict__ ids,
    const float* __restrict__ emb, bf16_t* __restrict__ X)
{
    const size_t g = (size_t)blockIdx.x * 256 + threadIdx.x;   // 4-elt groups
    const size_t e = g * 4;
    const int m = (int)(e >> 13);            // / (2*HS)
    const int c = (int)(e & 8191);
    const float* src = (c < CHS)
        ? emb + (size_t)ids[m] * CHS + c
        : hid + (size_t)m * CHS + (c - CHS);
    const float4 f = *(const float4*)src;
    X[e + 0] = f2b(f.x); X[e + 1] = f2b(f.y);
    X[e + 2] = f2b(f.z); X[e + 3] = f2b(f.w);
}

// ---------------------------------------------------------------------------
// RoPE in-place on q and k  (layout [B,S,NH,HD])
// ---------------------------------------------------------------------------
__global__ __launch_bounds__(256) void eagle_rope(bf16_t* __restrict__ q,
                                                  bf16_t* __restrict__ k)
{
    const int idx = blockIdx.x * 256 + threadIdx.x;     // B*S*NH*64 threads
    const int d = idx & 63;
    int t = idx >> 6;
    const int h = t & (CNH - 1); t >>= 5;
    const int s = t & (CS - 1);
    const int b = t >> 11;
    const float inv_freq = __powf(10000.0f, -(float)(2 * d) / (float)CHD);
    const float ang = (float)s * inv_freq;
    const float c = __cosf(ang), sn = __sinf(ang);
    const size_t base = (((size_t)(b * CS + s)) * CNH + h) * CHD;
    {
        float x1 = b2f(q[base + d]), x2 = b2f(q[base + d + 64]);
        q[base + d]      = f2b(x1 * c - x2 * sn);
        q[base + d + 64] = f2b(x2 * c + x1 * sn);
    }
    {
        float x1 = b2f(k[base + d]), x2 = b2f(k[base + d + 64]);
        k[base + d]      = f2b(x1 * c - x2 * sn);
        k[base + d + 64] = f2b(x2 * c + x1 * sn);
    }
}

// ---------------------------------------------------------------------------
// Flash attention: grid (B*NH, S/128), 8 waves/block, one 16-row Q tile per
// wave with online softmax. Per 32-key block: 8 score WMMAs + 8 PV WMMAs.
// V is consumed transposed ([B,H,D,S]) so PV B-fragments load contiguously.
// ---------------------------------------------------------------------------
__global__ __launch_bounds__(256) void eagle_flash_attn(
    const bf16_t* __restrict__ Q, const bf16_t* __restrict__ Km,
    const bf16_t* __restrict__ VT, bf16_t* __restrict__ CTX)
{
    constexpr int LD = CNH * CHD;           // 4096
    const int bh = blockIdx.x;              // 0..63
    const int b = bh >> 5, h = bh & (CNH - 1);
    const int wave = threadIdx.x >> 5;
    const int lane = threadIdx.x & 31;
    const int lrow = lane & 15;
    const int khalf = lane >> 4;
    const int koff = khalf << 3;
    const int q0 = blockIdx.y * 128 + wave * 16;

    const bf16_t* qb  = Q  + ((size_t)b * CS) * LD + (size_t)h * CHD;
    const bf16_t* kbp = Km + ((size_t)b * CS) * LD + (size_t)h * CHD;
    const bf16_t* vtb = VT + (size_t)bh * CHD * CS;   // [HD][S]

    __shared__ alignas(16) bf16_t pbuf[8][16 * 32];

    v16bf qf[4];
#pragma unroll
    for (int c = 0; c < 4; ++c)
        qf[c] = load_frag(qb + (size_t)(q0 + lrow) * LD + c * 32 + koff);

    v8f o[8];
#pragma unroll
    for (int i = 0; i < 8; ++i)
#pragma unroll
        for (int e = 0; e < 8; ++e) o[i][e] = 0.0f;

    float mrow[8], lsum[8];
#pragma unroll
    for (int r = 0; r < 8; ++r) { mrow[r] = -1e30f; lsum[r] = 0.0f; }

    const float scale = 0.08838834764831845f;   // 1/sqrt(128)

    for (int kb0 = 0; kb0 < q0 + 16; kb0 += 32) {
        // ---- scores: S = Q(16x128) . K^T  -> two 16x16 tiles (32 keys) ----
        v8f s0, s1;
#pragma unroll
        for (int e = 0; e < 8; ++e) { s0[e] = 0.0f; s1[e] = 0.0f; }
#pragma unroll
        for (int c = 0; c < 4; ++c) {
            v16bf k0 = load_frag(kbp + (size_t)(kb0 + lrow) * LD + c * 32 + koff);
            v16bf k1 = load_frag(kbp + (size_t)(kb0 + 16 + lrow) * LD + c * 32 + koff);
            s0 = __builtin_amdgcn_wmma_f32_16x16x32_bf16(false, qf[c], false, k0, (short)0, s0, false, false);
            s1 = __builtin_amdgcn_wmma_f32_16x16x32_bf16(false, qf[c], false, k1, (short)0, s1, false, false);
        }
        // ---- online softmax over this 32-key block ----
#pragma unroll
        for (int r = 0; r < 8; ++r) {
            const int qi = q0 + r + (khalf << 3);
            const int c0 = kb0 + lrow;
            const int c1 = kb0 + 16 + lrow;
            float v0 = (c0 <= qi) ? s0[r] * scale : -1e30f;
            float v1 = (c1 <= qi) ? s1[r] * scale : -1e30f;
            float mx = fmaxf(v0, v1);
            mx = fmaxf(mx, __shfl_xor(mx, 1, 32));
            mx = fmaxf(mx, __shfl_xor(mx, 2, 32));
            mx = fmaxf(mx, __shfl_xor(mx, 4, 32));
            mx = fmaxf(mx, __shfl_xor(mx, 8, 32));
            const float mnew = fmaxf(mrow[r], mx);
            const float corr = __expf(mrow[r] - mnew);
            const float p0 = __expf(v0 - mnew);
            const float p1 = __expf(v1 - mnew);
            float ps = p0 + p1;
            ps += __shfl_xor(ps, 1, 32);
            ps += __shfl_xor(ps, 2, 32);
            ps += __shfl_xor(ps, 4, 32);
            ps += __shfl_xor(ps, 8, 32);
            lsum[r] = lsum[r] * corr + ps;
            mrow[r] = mnew;
#pragma unroll
            for (int nt = 0; nt < 8; ++nt) o[nt][r] *= corr;
            // stash P (C-layout) into LDS for A-layout reload
            const int prow = r + (khalf << 3);
            pbuf[wave][prow * 32 + lrow]      = f2b(p0);
            pbuf[wave][prow * 32 + 16 + lrow] = f2b(p1);
        }
        asm volatile("s_wait_dscnt 0" ::: "memory");
        // ---- PV: O += P(16x32) . V(32x128) ----
        const v16bf pf = load_frag(&pbuf[wave][(size_t)lrow * 32 + koff]);
#pragma unroll
        for (int nt = 0; nt < 8; ++nt) {
            v16bf vf = load_frag(vtb + (size_t)(nt * 16 + lrow) * CS + kb0 + koff);
            o[nt] = __builtin_amdgcn_wmma_f32_16x16x32_bf16(false, pf, false, vf, (short)0, o[nt], false, false);
        }
    }

    // ---- normalize + store ctx at [B,S,NH*HD] ----
#pragma unroll
    for (int r = 0; r < 8; ++r) {
        const float inv = 1.0f / lsum[r];
        const int s = q0 + r + (khalf << 3);
        const size_t base = ((size_t)(b * CS + s)) * LD + (size_t)h * CHD;
#pragma unroll
        for (int nt = 0; nt < 8; ++nt)
            CTX[base + nt * 16 + lrow] = f2b(o[nt][r] * inv);
    }
}

// ---------------------------------------------------------------------------
// RMSNorm: one block per row of [4096, 4096] f32 -> bf16
// ---------------------------------------------------------------------------
__global__ __launch_bounds__(256) void eagle_rmsnorm(
    const float* __restrict__ X, const float* __restrict__ W,
    bf16_t* __restrict__ Y)
{
    const int row = blockIdx.x;
    const float* x = X + (size_t)row * CHS;
    float ss = 0.0f;
    for (int i = threadIdx.x; i < CHS; i += 256) { const float v = x[i]; ss += v * v; }
#pragma unroll
    for (int off = 16; off; off >>= 1) ss += __shfl_xor(ss, off, 32);
    __shared__ float red[8];
    __shared__ float sinv;
    const int wave = threadIdx.x >> 5, lane = threadIdx.x & 31;
    if (lane == 0) red[wave] = ss;
    __syncthreads();
    if (threadIdx.x == 0) {
        float t = 0.0f;
        for (int i = 0; i < 8; ++i) t += red[i];
        sinv = rsqrtf(t * (1.0f / (float)CHS) + 1e-6f);
    }
    __syncthreads();
    const float inv = sinv;
    for (int i = threadIdx.x; i < CHS; i += 256)
        Y[(size_t)row * CHS + i] = f2b(W[i] * x[i] * inv);
}

// ---------------------------------------------------------------------------
// act = silu(gate) * up   (bf16 in / bf16 out, in-place into gate allowed)
// ---------------------------------------------------------------------------
__global__ __launch_bounds__(256) void eagle_silu_mul(
    const bf16_t* __restrict__ g, const bf16_t* __restrict__ u, bf16_t* a)
{
    const size_t i = (size_t)blockIdx.x * 256 + threadIdx.x;
    const float gv = b2f(g[i]);
    const float uv = b2f(u[i]);
    const float s = gv / (1.0f + __expf(-gv));
    a[i] = f2b(s * uv);
}

// ---------------------------------------------------------------------------
// Launcher
// ---------------------------------------------------------------------------
extern "C" void kernel_launch(void* const* d_in, const int* in_sizes, int n_in,
                              void* d_out, int out_size, void* d_ws, size_t ws_size,
                              hipStream_t stream) {
    (void)in_sizes; (void)n_in; (void)out_size; (void)ws_size;

    const float* hidden = (const float*)d_in[0];
    const int*   ids    = (const int*)d_in[1];
    (void)d_in[2];                                  // position_ids == arange(S)
    const float* embed  = (const float*)d_in[3];
    const float* fc_w   = (const float*)d_in[4];
    const float* fc_b   = (const float*)d_in[5];
    const float* q_w    = (const float*)d_in[6];
    const float* k_w    = (const float*)d_in[7];
    const float* v_w    = (const float*)d_in[8];
    const float* o_w    = (const float*)d_in[9];
    const float* gate_w = (const float*)d_in[10];
    const float* up_w   = (const float*)d_in[11];
    const float* down_w = (const float*)d_in[12];
    const float* ln_w   = (const float*)d_in[13];
    float* out = (float*)d_out;

    // ---- workspace layout (lifetime-based reuse), ~853 MB total ----
    auto al = [](size_t x) { return (x + 255) & ~(size_t)255; };
    char* ws = (char*)d_ws;
    const size_t szA   = al((size_t)CM * CINTER * 2);     // region A: xcat/ctx/gate
    const size_t szUp  = szA;
    const size_t szH   = al((size_t)CM * CHS * 4);
    const size_t szBf  = al((size_t)CM * CHS * 2);
    const size_t szFCW = al((size_t)CHS * (2 * CHS) * 2);
    const size_t szPW  = al((size_t)CHS * CHS * 2);
    const size_t szIW  = al((size_t)CINTER * CHS * 2);

    size_t off = 0;
    const size_t oA   = off; off += szA;
    const size_t oUp  = off; off += szUp;
    const size_t oH   = off; off += szH;
    const size_t oQ   = off; off += szBf;
    const size_t oK   = off; off += szBf;
    const size_t oV   = off; off += szBf;
    const size_t oHb  = off; off += szBf;
    const size_t oFCW = off; off += szFCW;
    const size_t oQW  = off; off += szPW;
    const size_t oKW  = off; off += szPW;
    const size_t oVW  = off; off += szPW;
    const size_t oOW  = off; off += szPW;
    const size_t oGW  = off; off += szIW;
    const size_t oUW  = off; off += szIW;
    const size_t oDW  = off; off += szIW;

    bf16_t* XCAT = (bf16_t*)(ws + oA);     // [4096, 8192] bf16
    bf16_t* CTX  = (bf16_t*)(ws + oA);     // [4096, 4096] bf16 (after XCAT dies)
    bf16_t* GATE = (bf16_t*)(ws + oA);     // [4096,11008] bf16 (after CTX dies)
    bf16_t* UPB  = (bf16_t*)(ws + oUp);    // [4096,11008] bf16
    float*  H    = (float*) (ws + oH);     // [4096, 4096] f32 residual
    bf16_t* Qb   = (bf16_t*)(ws + oQ);     // [B,S,NH,HD] bf16
    bf16_t* Kb   = (bf16_t*)(ws + oK);     // [B,S,NH,HD] bf16
    bf16_t* HN   = (bf16_t*)(ws + oK);     // reuses K after attention
    bf16_t* VTb  = (bf16_t*)(ws + oV);     // [B,NH,HD,S] bf16 (transposed V)
    bf16_t* Hb   = (bf16_t*)(ws + oHb);    // bf16 copy of h
    bf16_t* FCW  = (bf16_t*)(ws + oFCW);
    bf16_t* QW   = (bf16_t*)(ws + oQW);
    bf16_t* KW   = (bf16_t*)(ws + oKW);
    bf16_t* VW   = (bf16_t*)(ws + oVW);
    bf16_t* OW   = (bf16_t*)(ws + oOW);
    bf16_t* GW   = (bf16_t*)(ws + oGW);
    bf16_t* UW   = (bf16_t*)(ws + oUW);
    bf16_t* DW   = (bf16_t*)(ws + oDW);

    const dim3 blk(256);
    auto cvt = [&](const float* s, bf16_t* d, size_t n) {
        eagle_f32_to_bf16<<<(unsigned)(n / 2048), blk, 0, stream>>>(s, d);
    };

    // 0) one-shot weight conversion fp32 -> bf16 (bandwidth-bound, ~65us)
    cvt(fc_w,   FCW, (size_t)CHS * 2 * CHS);
    cvt(q_w,    QW,  (size_t)CHS * CHS);
    cvt(k_w,    KW,  (size_t)CHS * CHS);
    cvt(v_w,    VW,  (size_t)CHS * CHS);
    cvt(o_w,    OW,  (size_t)CHS * CHS);
    cvt(gate_w, GW,  (size_t)CINTER * CHS);
    cvt(up_w,   UW,  (size_t)CINTER * CHS);
    cvt(down_w, DW,  (size_t)CHS * CINTER);

    // 1) Xcat = concat(embed[ids], hidden) -> bf16
    eagle_embed_concat<<<32768, blk, 0, stream>>>(hidden, ids, embed, XCAT);

    // 2) h = Xcat @ fc_w^T + fc_b   [4096,4096] f32
    gemm_wmma<float, false, true, false>
        <<<dim3(32, 16), blk, 0, stream>>>(XCAT, FCW, nullptr, fc_b, H, CM, CHS, 2 * CHS);

    // 2b) bf16 copy of h for the QKV projections
    cvt(H, Hb, (size_t)CM * CHS);

    // 3) q,k,v projections (v stored transposed per-head)
    gemm_wmma<bf16_t, false, false, false>
        <<<dim3(32, 16), blk, 0, stream>>>(Hb, QW, nullptr, nullptr, Qb, CM, CHS, CHS);
    gemm_wmma<bf16_t, false, false, false>
        <<<dim3(32, 16), blk, 0, stream>>>(Hb, KW, nullptr, nullptr, Kb, CM, CHS, CHS);
    gemm_wmma<bf16_t, false, false, true>
        <<<dim3(32, 16), blk, 0, stream>>>(Hb, VW, nullptr, nullptr, VTb, CM, CHS, CHS);

    // 4) RoPE in-place on q,k
    eagle_rope<<<32768, blk, 0, stream>>>(Qb, Kb);

    // 5) ctx = causal_flash_attention(q,k,v)
    eagle_flash_attn<<<dim3(CB * CNH, CS / 128), blk, 0, stream>>>(Qb, Kb, VTb, CTX);

    // 6) h += ctx @ o_w^T   (in-place residual add)
    gemm_wmma<float, true, false, false>
        <<<dim3(32, 16), blk, 0, stream>>>(CTX, OW, H, nullptr, H, CM, CHS, CHS);

    // 7) hn = rmsnorm(h) * post_ln_w  -> bf16
    eagle_rmsnorm<<<CM, blk, 0, stream>>>(H, ln_w, HN);

    // 8) gate / up projections -> bf16 [4096, 11008]
    gemm_wmma<bf16_t, false, false, false>
        <<<dim3(32, 43), blk, 0, stream>>>(HN, GW, nullptr, nullptr, GATE, CM, CINTER, CHS);
    gemm_wmma<bf16_t, false, false, false>
        <<<dim3(32, 43), blk, 0, stream>>>(HN, UW, nullptr, nullptr, UPB, CM, CINTER, CHS);

    // 9) act = silu(gate) * up  (in-place into gate region)
    eagle_silu_mul<<<(unsigned)(((size_t)CM * CINTER) / 256), blk, 0, stream>>>(GATE, UPB, GATE);

    // 10) out = h + act @ down_w^T   [4096,4096] f32
    gemm_wmma<float, true, false, false>
        <<<dim3(32, 16), blk, 0, stream>>>(GATE, DW, H, nullptr, out, CM, CHS, CINTER);
}
